// EnhancedGNN_24232205484082
// MI455X (gfx1250) — matrix operand order repaired and checked by reading, not compiled, not run on hardware
//
#include <hip/hip_runtime.h>
#include <hip/hip_bf16.h>
#include <math.h>

#define HID 128
#define EPS 1e-5f

typedef __attribute__((ext_vector_type(16))) _Float16 v16h;
typedef __attribute__((ext_vector_type(8)))  _Float16 half8;
typedef __attribute__((ext_vector_type(8)))  float    v8f;

__device__ __forceinline__ float wred(float v) {
#pragma unroll
  for (int i = 16; i > 0; i >>= 1) v += __shfl_xor(v, i, 32);
  return v;
}

// ---------------------------------------------------------------------------
// Weight prep: transpose W[k][n] -> Wt[n][k] and convert to f16.
// Wt layout: [W1t | W2t | W3t | fc1t], each row has 128 k-values.
// ---------------------------------------------------------------------------
__global__ void prep_weights(const float* __restrict__ W1,
                             const float* __restrict__ W2,
                             const float* __restrict__ W3,
                             const float* __restrict__ fc1W,
                             _Float16* __restrict__ Wt) {
  int idx = blockIdx.x * blockDim.x + threadIdx.x;
  const int NW = 3 * 16384 + 32 * 128;
  if (idx >= NW) return;
  if (idx < 3 * 16384) {
    const float* W = (idx < 16384) ? W1 : (idx < 32768 ? W2 : W3);
    int r = idx & 16383;
    int n = r >> 7, k = r & 127;
    Wt[idx] = (_Float16)W[k * 128 + n];
  } else {
    int r = idx - 3 * 16384;
    int n = r >> 7, k = r & 127;
    Wt[idx] = (_Float16)fc1W[k * 32 + n];
  }
}

// ---------------------------------------------------------------------------
// Degree / norm build
// ---------------------------------------------------------------------------
__global__ void k_init_deg(float* __restrict__ deg, int n) {
  int i = blockIdx.x * blockDim.x + threadIdx.x;
  if (i < n) deg[i] = 1.0f;  // self-loop
}
__global__ void k_deg_accum(const int* __restrict__ ei, float* __restrict__ deg, int E) {
  int e = blockIdx.x * blockDim.x + threadIdx.x;
  if (e < E) atomicAdd(&deg[ei[E + e]], 1.0f);  // col = edge_index[1]
}
__global__ void k_deg_inv(float* __restrict__ deg, int n) {
  int i = blockIdx.x * blockDim.x + threadIdx.x;
  if (i < n) deg[i] = rsqrtf(deg[i]);
}
__global__ void k_norm(const int* __restrict__ ei, const float* __restrict__ dinv,
                       float* __restrict__ nrm, int E, int n) {
  int e = blockIdx.x * blockDim.x + threadIdx.x;
  if (e >= E + n) return;
  if (e < E) nrm[e] = dinv[ei[e]] * dinv[ei[E + e]];
  else { float d = dinv[e - E]; nrm[e] = d * d; }
}

// ---------------------------------------------------------------------------
// Embed: h = layernorm(relu(x @ coord_W + coord_b)); one wave per node.
// ---------------------------------------------------------------------------
__global__ __launch_bounds__(256)
void embed_ln(const float* __restrict__ x, const float* __restrict__ cW,
              const float* __restrict__ cB, const float* __restrict__ lg,
              const float* __restrict__ lb, float* __restrict__ h,
              _Float16* __restrict__ hf, int n) {
  int node = (int)((blockIdx.x * blockDim.x + threadIdx.x) >> 5);
  int lane = threadIdx.x & 31;
  if (node >= n) return;
  float x0 = x[node * 2 + 0], x1 = x[node * 2 + 1];
  int c0 = lane * 4;
  float t[4];
#pragma unroll
  for (int j = 0; j < 4; ++j) {
    int c = c0 + j;
    float v = fmaf(x0, cW[c], fmaf(x1, cW[128 + c], cB[c]));
    t[j] = fmaxf(v, 0.f);
  }
  float s  = wred(t[0] + t[1] + t[2] + t[3]);
  float sq = wred(t[0]*t[0] + t[1]*t[1] + t[2]*t[2] + t[3]*t[3]);
  float m   = s * (1.f / 128.f);
  float var = sq * (1.f / 128.f) - m * m;
  float inv = rsqrtf(var + EPS);
#pragma unroll
  for (int j = 0; j < 4; ++j) {
    int c = c0 + j;
    float o = (t[j] - m) * inv * lg[c] + lb[c];
    h[(size_t)node * HID + c]  = o;
    hf[(size_t)node * HID + c] = (_Float16)o;
  }
}

// ---------------------------------------------------------------------------
// WMMA GEMM: C[nrows x 16*NT] = A[nrows x 128] * Bt^T, A/Bt f16, C f32.
// Bt is column-major weights: Bt[n][k], 128 k per row.
// Bt is staged once per block into LDS via async global->LDS copies
// (ASYNCcnt path), with a padded row stride to avoid bank conflicts.
// One wave per 16-row strip; NT 16-col tiles; K=128 in four 32-steps.
// ---------------------------------------------------------------------------
template <int NT>
__global__ __launch_bounds__(128)
void gemm_wmma_f16(const _Float16* __restrict__ A, const _Float16* __restrict__ Bt,
                   float* __restrict__ C, int nrows, int ldc) {
  constexpr int SBROW = 136;                 // padded LDS row stride (halves)
  __shared__ _Float16 sB[NT * 16 * SBROW];

  const int lane = threadIdx.x & 31;
  const int wave = threadIdx.x >> 5;

  // ---- cooperative async stage of Bt (NT*16 rows x 128 halves) into LDS ----
  {
    const int t = threadIdx.x;
#pragma unroll
    for (int i = 0; i < NT * 2; ++i) {
      int idx = t + i * 128;                 // 16-byte chunk id
      int n = idx >> 4;                      // Bt row (output column)
      int c = idx & 15;                      // 16B chunk within row
      unsigned lds_off = (unsigned)(uintptr_t)(&sB[n * SBROW + c * 8]);
      const _Float16* g = Bt + (size_t)n * 128 + c * 8;
      asm volatile("global_load_async_to_lds_b128 %0, %1, off"
                   :: "v"(lds_off), "v"((unsigned long long)(uintptr_t)g)
                   : "memory");
    }
    asm volatile("s_wait_asynccnt 0x0" ::: "memory");
  }
  __syncthreads();                            // all waves staged before compute

  const int rt = blockIdx.x * 4 + wave;
  if (rt * 16 >= nrows) return;              // wave-uniform: EXEC stays full
  const int mr   = lane & 15;
  const int hi   = lane >> 4;                // 0 or 1
  const int kloA = hi * 8;                   // A: K offset 0 or 8
  const int kloB = hi * 8;                   // B: K offset 0 or 8
  const _Float16* arow = A + (size_t)(rt * 16 + mr) * 128;

  v8f acc[NT];
  const v8f zero = {0.f, 0.f, 0.f, 0.f, 0.f, 0.f, 0.f, 0.f};
#pragma unroll
  for (int j = 0; j < NT; ++j) acc[j] = zero;

#pragma unroll
  for (int kb = 0; kb < 128; kb += 32) {
    union { v16h v; half8 h[2]; } a;
    a.h[0] = *(const half8*)(arow + kb + kloA);
    a.h[1] = *(const half8*)(arow + kb + 16 + kloA);
#pragma unroll
    for (int j = 0; j < NT; ++j) {
      const _Float16* bcol = &sB[(j * 16 + mr) * SBROW + kb + kloB];
      union { v16h v; half8 h[2]; } b;
      b.h[0] = *(const half8*)(bcol);
      b.h[1] = *(const half8*)(bcol + 16);
      acc[j] = __builtin_amdgcn_wmma_f32_16x16x32_f16(
          false, a.v, false, b.v, (short)0, acc[j], false, false);
    }
  }
  const int rbase = rt * 16 + hi * 8;
#pragma unroll
  for (int j = 0; j < NT; ++j) {
#pragma unroll
    for (int v = 0; v < 8; ++v) {
      C[(size_t)(rbase + v) * ldc + j * 16 + mr] = acc[j][v];
    }
  }
}

// ---------------------------------------------------------------------------
// Edge scatter: agg[col] += norm[e] * xw[row]; one wave per edge, 4 f32/lane.
// ---------------------------------------------------------------------------
__global__ __launch_bounds__(256)
void k_scatter(const float* __restrict__ xw, const float* __restrict__ nrm,
               const int* __restrict__ ei, float* __restrict__ agg, int E, int n) {
  long long gw = (long long)((blockIdx.x * (long long)blockDim.x + threadIdx.x) >> 5);
  int lane = threadIdx.x & 31;
  long long total = (long long)E + n;
  if (gw >= total) return;
  int r, c;
  if (gw < E) { r = ei[gw]; c = ei[E + gw]; }
  else        { r = c = (int)(gw - E); }
  float w = nrm[gw];
  const float4 v = ((const float4*)(xw + (size_t)r * HID))[lane];
  float* dst = agg + (size_t)c * HID + lane * 4;
  atomicAdd(dst + 0, v.x * w);
  atomicAdd(dst + 1, v.y * w);
  atomicAdd(dst + 2, v.z * w);
  atomicAdd(dst + 3, v.w * w);
}

// ---------------------------------------------------------------------------
// BN stats: y = relu(agg + bias); accumulate per-channel sum / sumsq.
// ---------------------------------------------------------------------------
__global__ __launch_bounds__(256)
void k_bn_stats(const float* __restrict__ agg, const float* __restrict__ bias,
                float* __restrict__ y, float* __restrict__ csum,
                float* __restrict__ csq, int n) {
  __shared__ float ss[HID], sq[HID];
  if (threadIdx.x < HID) { ss[threadIdx.x] = 0.f; sq[threadIdx.x] = 0.f; }
  __syncthreads();
  size_t total = (size_t)n * HID;
  size_t stride = (size_t)gridDim.x * blockDim.x;
  for (size_t i = (size_t)blockIdx.x * blockDim.x + threadIdx.x; i < total; i += stride) {
    int c = (int)(i & (HID - 1));
    float v = fmaxf(agg[i] + bias[c], 0.f);
    y[i] = v;
    atomicAdd(&ss[c], v);
    atomicAdd(&sq[c], v * v);
  }
  __syncthreads();
  if (threadIdx.x < HID) {
    atomicAdd(&csum[threadIdx.x], ss[threadIdx.x]);
    atomicAdd(&csq[threadIdx.x],  sq[threadIdx.x]);
  }
}

__global__ void k_bn_finalize(const float* __restrict__ csum, const float* __restrict__ csq,
                              const float* __restrict__ g, const float* __restrict__ b,
                              float* __restrict__ scale, float* __restrict__ shift, int n) {
  int c = threadIdx.x;
  if (c < HID) {
    float invn = 1.f / (float)n;
    float m = csum[c] * invn;
    float v = csq[c] * invn - m * m;
    float s = g[c] * rsqrtf(v + EPS);
    scale[c] = s;
    shift[c] = b[c] - m * s;
  }
}

// ---------------------------------------------------------------------------
// BN apply + residual + instance-norm; writes new h (f32) and hf (f16).
// One wave per node.
// ---------------------------------------------------------------------------
__global__ __launch_bounds__(256)
void k_post_norm(const float* __restrict__ y, const float* __restrict__ scale,
                 const float* __restrict__ shift, float* __restrict__ h,
                 _Float16* __restrict__ hf, int n) {
  int node = (int)((blockIdx.x * blockDim.x + threadIdx.x) >> 5);
  int lane = threadIdx.x & 31;
  if (node >= n) return;
  const float4 yv = ((const float4*)(y + (size_t)node * HID))[lane];
  const float4 id = ((const float4*)(h + (size_t)node * HID))[lane];
  const float4 sc = ((const float4*)scale)[lane];
  const float4 sh = ((const float4*)shift)[lane];
  float t[4];
  t[0] = fmaf(yv.x, sc.x, sh.x) + id.x;
  t[1] = fmaf(yv.y, sc.y, sh.y) + id.y;
  t[2] = fmaf(yv.z, sc.z, sh.z) + id.z;
  t[3] = fmaf(yv.w, sc.w, sh.w) + id.w;
  float s  = wred(t[0] + t[1] + t[2] + t[3]);
  float sq = wred(t[0]*t[0] + t[1]*t[1] + t[2]*t[2] + t[3]*t[3]);
  float m   = s * (1.f / 128.f);
  float var = sq * (1.f / 128.f) - m * m;
  float inv = rsqrtf(var + EPS);
#pragma unroll
  for (int j = 0; j < 4; ++j) {
    float o = (t[j] - m) * inv;
    h[(size_t)node * HID + lane * 4 + j]  = o;
    hf[(size_t)node * HID + lane * 4 + j] = (_Float16)o;
  }
}

// ---------------------------------------------------------------------------
// Head: v = relu(fc1raw + fc1_b); out = tanh(v @ fc2_W + fc2_b). Wave/node.
// ---------------------------------------------------------------------------
__global__ __launch_bounds__(256)
void k_head(const float* __restrict__ fc1raw, const float* __restrict__ fc1b,
            const float* __restrict__ fc2W, const float* __restrict__ fc2b,
            float* __restrict__ out, int n) {
  int node = (int)((blockIdx.x * blockDim.x + threadIdx.x) >> 5);
  int lane = threadIdx.x & 31;
  if (node >= n) return;
  float v = fmaxf(fc1raw[(size_t)node * 32 + lane] + fc1b[lane], 0.f);
  float a0 = wred(v * fc2W[lane * 2 + 0]);
  float a1 = wred(v * fc2W[lane * 2 + 1]);
  if (lane == 0) {
    out[(size_t)node * 2 + 0] = tanhf(a0 + fc2b[0]);
    out[(size_t)node * 2 + 1] = tanhf(a1 + fc2b[1]);
  }
}

// ---------------------------------------------------------------------------
extern "C" void kernel_launch(void* const* d_in, const int* in_sizes, int n_in,
                              void* d_out, int out_size, void* d_ws, size_t ws_size,
                              hipStream_t stream) {
  const float* x      = (const float*)d_in[0];
  const int*   ei     = (const int*)  d_in[1];
  const float* coordW = (const float*)d_in[2];
  const float* coordB = (const float*)d_in[3];
  const float* lng    = (const float*)d_in[4];
  const float* lnb    = (const float*)d_in[5];
  const float* bng    = (const float*)d_in[6];
  const float* bnb    = (const float*)d_in[7];
  const float* Ws[3]  = {(const float*)d_in[8], (const float*)d_in[10], (const float*)d_in[12]};
  const float* bs[3]  = {(const float*)d_in[9], (const float*)d_in[11], (const float*)d_in[13]};
  const float* fc1W   = (const float*)d_in[14];
  const float* fc1b   = (const float*)d_in[15];
  const float* fc2W   = (const float*)d_in[16];
  const float* fc2b   = (const float*)d_in[17];

  const int N = in_sizes[0] / 2;
  const int E = in_sizes[1] / 2;

  // workspace layout (256B aligned slices)
  uint8_t* base = (uint8_t*)d_ws;
  size_t off = 0;
  auto take = [&](size_t bytes) -> void* {
    void* p = base + off;
    off += (bytes + 255) & ~(size_t)255;
    return p;
  };
  float*     deg   = (float*)take((size_t)N * 4);                 // -> dinv in place
  float*     nrm   = (float*)take((size_t)(E + N) * 4);
  float*     h     = (float*)take((size_t)N * HID * 4);
  float*     xw    = (float*)take((size_t)N * HID * 4);           // gemm out, reused as y
  float*     agg   = (float*)take((size_t)N * HID * 4);
  _Float16*  hf    = (_Float16*)take((size_t)N * HID * 2);
  _Float16*  Wt    = (_Float16*)take((size_t)(3 * 16384 + 32 * 128) * 2);
  float*     csum  = (float*)take(HID * 4);
  float*     csq   = (float*)take(HID * 4);
  float*     scale = (float*)take(HID * 4);
  float*     shift = (float*)take(HID * 4);

  const int NWELT = 3 * 16384 + 32 * 128;
  prep_weights<<<(NWELT + 255) / 256, 256, 0, stream>>>(Ws[0], Ws[1], Ws[2], fc1W, Wt);

  k_init_deg <<<(N + 255) / 256, 256, 0, stream>>>(deg, N);
  k_deg_accum<<<(E + 255) / 256, 256, 0, stream>>>(ei, deg, E);
  k_deg_inv  <<<(N + 255) / 256, 256, 0, stream>>>(deg, N);
  k_norm     <<<(E + N + 255) / 256, 256, 0, stream>>>(ei, deg, nrm, E, N);

  const int nodeWaveBlocks = (N * 32 + 255) / 256;
  embed_ln<<<nodeWaveBlocks, 256, 0, stream>>>(x, coordW, coordB, lng, lnb, h, hf, N);

  const int rowTiles   = (N + 15) / 16;
  const int gemmBlocks = (rowTiles + 3) / 4;
  const long long edgeWaves  = (long long)E + N;
  const int scatterBlocks = (int)((edgeWaves * 32 + 255) / 256);

  for (int l = 0; l < 3; ++l) {
    gemm_wmma_f16<8><<<gemmBlocks, 128, 0, stream>>>(hf, Wt + (size_t)l * 16384, xw, N, HID);
    hipMemsetAsync(agg, 0, (size_t)N * HID * 4, stream);
    k_scatter<<<scatterBlocks, 256, 0, stream>>>(xw, nrm, ei, agg, E, N);
    hipMemsetAsync(csum, 0, HID * 4, stream);
    hipMemsetAsync(csq,  0, HID * 4, stream);
    k_bn_stats<<<1024, 256, 0, stream>>>(agg, bs[l], xw, csum, csq, N);  // y reuses xw
    k_bn_finalize<<<1, 128, 0, stream>>>(csum, csq, bng, bnb, scale, shift, N);
    k_post_norm<<<nodeWaveBlocks, 256, 0, stream>>>(xw, scale, shift, h, hf, N);
  }

  gemm_wmma_f16<2><<<gemmBlocks, 128, 0, stream>>>(hf, Wt + (size_t)3 * 16384, xw, N, 32);
  k_head<<<nodeWaveBlocks, 256, 0, stream>>>(xw, fc1b, fc2W, fc2b, (float*)d_out, N);
}
